// TransformerBlock_29094108463548
// MI455X (gfx1250) — matrix operand order repaired and checked by reading, not compiled, not run on hardware
//
#include <hip/hip_runtime.h>
#include <stdint.h>

// ---------------- problem constants ----------------
constexpr int kB  = 4;
constexpr int kT  = 2048;
constexpr int kC  = 1024;
constexpr int kH  = 16;
constexpr int kDH = 64;
constexpr int kFF = 4096;
constexpr int kBT = kB * kT;

typedef __attribute__((ext_vector_type(16))) __bf16    v16bf;
typedef __attribute__((ext_vector_type(8)))  float     v8f;
typedef __attribute__((ext_vector_type(8)))  uint32_t  v8u;

#if defined(__has_builtin)
#if __has_builtin(__builtin_amdgcn_global_load_async_to_lds_b128)
#define HAS_ASYNC_LDS 1
#endif
#endif
#ifndef HAS_ASYNC_LDS
#define HAS_ASYNC_LDS 0
#endif

#if HAS_ASYNC_LDS
typedef int __attribute__((ext_vector_type(4))) v4i;
typedef v4i __attribute__((address_space(1)))* as1_v4i;
typedef v4i __attribute__((address_space(3)))* as3_v4i;
#if __has_builtin(__builtin_amdgcn_s_wait_asynccnt)
#define WAIT_ASYNC(n) __builtin_amdgcn_s_wait_asynccnt(n)
#else
#define WAIT_ASYNC(n) asm volatile("s_wait_asynccnt %0" ::"n"(n) : "memory")
#endif
#endif

// fp32 -> bf16 with round-to-nearest-even
__device__ __forceinline__ unsigned short f2bf(float f) {
  uint32_t u = __builtin_bit_cast(uint32_t, f);
  u = (u + 0x7FFFu + ((u >> 16) & 1u)) >> 16;
  return (unsigned short)u;
}

// ---- WMMA fragment loaders (contiguous K order, same for A and B, wave32) ----
// A-matrix 16x32 bf16, row-major source, leading dim `ld` (elements).
// lane<16 holds K in {0..15}; lane>=16 holds {16..31}.
__device__ __forceinline__ v16bf load_a16x32(const unsigned short* base, int ld,
                                             int row0, int k0, int lane) {
  int row = row0 + (lane & 15);
  int kh  = (lane >> 4) * 16;
  const unsigned short* p = base + (size_t)row * ld + k0 + kh;
  const uint32_t* p0 = (const uint32_t*)p;          // K = kh .. kh+7
  const uint32_t* p1 = (const uint32_t*)(p + 8);    // K = kh+8 .. kh+15
  v8u w;
#pragma unroll
  for (int j = 0; j < 4; ++j) { w[j] = p0[j]; w[4 + j] = p1[j]; }
  return __builtin_bit_cast(v16bf, w);
}

// B-matrix 32x16 bf16 from transposed (column-major) source Bt[N][K],
// leading dim `ldk`. Lane holds column N=n0+(lane&15), 16 contiguous K.
__device__ __forceinline__ v16bf load_b32x16(const unsigned short* bt, int ldk,
                                             int n0, int k0, int lane) {
  int n  = n0 + (lane & 15);
  int kb = k0 + (lane >> 4) * 16;
  const uint32_t* p = (const uint32_t*)(bt + (size_t)n * ldk + kb);
  v8u w;
#pragma unroll
  for (int j = 0; j < 8; ++j) w[j] = p[j];
  return __builtin_bit_cast(v16bf, w);
}

__device__ __forceinline__ v8f wmma_bf16(v16bf a, v16bf b, v8f c) {
  return __builtin_amdgcn_wmma_f32_16x16x32_bf16(false, a, false, b,
                                                 (short)0, c, false, false);
}

// ---------------- weight transpose + bf16 convert ----------------
// W[batch][K][N] f32  ->  Wt[batch][N][K] bf16
__global__ void k_transpose_bf16(const float* __restrict__ W,
                                 unsigned short* __restrict__ Wt,
                                 int K, int N, long total) {
  long i = (long)blockIdx.x * blockDim.x + threadIdx.x;
  if (i >= total) return;
  long per = (long)K * N;
  long b = i / per, r = i % per;
  int kk = (int)(r / N), nn = (int)(r % N);
  Wt[b * per + (long)nn * K + kk] = f2bf(W[i]);
}

// ---------------- layernorm -> bf16 ----------------
__global__ void k_layernorm(const float* __restrict__ x,
                            const float* __restrict__ g,
                            const float* __restrict__ be,
                            unsigned short* __restrict__ y) {
  int row = blockIdx.x, tid = threadIdx.x;
  const float* xr = x + (size_t)row * kC;
  __shared__ float s1[256], s2[256];
  float a = 0.f, b = 0.f;
  for (int i = tid; i < kC; i += 256) { float v = xr[i]; a += v; b += v * v; }
  s1[tid] = a; s2[tid] = b; __syncthreads();
  for (int off = 128; off > 0; off >>= 1) {
    if (tid < off) { s1[tid] += s1[tid + off]; s2[tid] += s2[tid + off]; }
    __syncthreads();
  }
  float mu = s1[0] / kC;
  float var = s2[0] / kC - mu * mu;
  float rs = rsqrtf(var + 1e-5f);
  for (int i = tid; i < kC; i += 256)
    y[(size_t)row * kC + i] = f2bf((xr[i] - mu) * rs * g[i] + be[i]);
}

// ---------------- QKV projection (per head) ----------------
// hn [BT, C] bf16 ; Wt [H][DH][C] bf16 (pre-transposed)
// q,k -> [B,H,T,DH] bf16 ; v -> [B,H,DH,T] bf16 (transposed for PV WMMA)
__global__ void k_qkv(const unsigned short* __restrict__ hn,
                      const unsigned short* __restrict__ Wq_t,
                      const unsigned short* __restrict__ Wk_t,
                      const unsigned short* __restrict__ Wv_t,
                      unsigned short* __restrict__ qb,
                      unsigned short* __restrict__ kb,
                      unsigned short* __restrict__ vtb) {
  int lane = threadIdx.x & 31, wave = threadIdx.x >> 5;
  int row0 = blockIdx.x * 128 + wave * 32;
  int h = blockIdx.y, which = blockIdx.z;
  const unsigned short* Wt =
      (which == 0 ? Wq_t : which == 1 ? Wk_t : Wv_t) + (size_t)h * kDH * kC;
  v8f acc[2][4];
#pragma unroll
  for (int mt = 0; mt < 2; ++mt)
#pragma unroll
    for (int nt = 0; nt < 4; ++nt)
#pragma unroll
      for (int r = 0; r < 8; ++r) acc[mt][nt][r] = 0.f;

  for (int k0 = 0; k0 < kC; k0 += 32) {
    v16bf a0 = load_a16x32(hn, kC, row0,      k0, lane);
    v16bf a1 = load_a16x32(hn, kC, row0 + 16, k0, lane);
#pragma unroll
    for (int nt = 0; nt < 4; ++nt) {
      v16bf b = load_b32x16(Wt, kC, nt * 16, k0, lane);
      acc[0][nt] = wmma_bf16(a0, b, acc[0][nt]);
      acc[1][nt] = wmma_bf16(a1, b, acc[1][nt]);
    }
  }
  int ncl = lane & 15, hf = lane >> 4;
#pragma unroll
  for (int mt = 0; mt < 2; ++mt)
#pragma unroll
    for (int nt = 0; nt < 4; ++nt)
#pragma unroll
      for (int r = 0; r < 8; ++r) {
        int m = row0 + mt * 16 + r + hf * 8;
        int bidx = m >> 11, t = m & (kT - 1);
        int d = nt * 16 + ncl;
        unsigned short bv = f2bf(acc[mt][nt][r]);
        if (which == 0)
          qb[((((size_t)bidx * kH + h) * kT + t) * kDH) + d] = bv;
        else if (which == 1)
          kb[((((size_t)bidx * kH + h) * kT + t) * kDH) + d] = bv;
        else
          vtb[((((size_t)bidx * kH + h) * kDH + d) * kT) + t] = bv;
      }
}

// ---------------- causal flash attention ----------------
// q,k [B,H,T,DH] bf16 ; vt [B,H,DH,T] bf16 ; out: attn [BT, C] bf16
// NOTE: waves have differing trip counts -> no __syncthreads in the loop.
// Per-wave LDS slice + same-wave in-order DS ops make it safe.
__global__ void k_attn(const unsigned short* __restrict__ q,
                       const unsigned short* __restrict__ k,
                       const unsigned short* __restrict__ vt,
                       unsigned short* __restrict__ attn_out) {
  constexpr float kScale = 0.125f;  // DH^-0.5
  int lane = threadIdx.x & 31, wave = threadIdx.x >> 5;
  int bh = blockIdx.y, b = bh / kH, h = bh % kH;
  int qt0 = (blockIdx.x * 4 + wave) * 16;
  const unsigned short* Q  = q  + (size_t)bh * kT * kDH;
  const unsigned short* Kp = k  + (size_t)bh * kT * kDH;
  const unsigned short* Vt = vt + (size_t)bh * kDH * kT;

  __shared__ unsigned short Pt[4][16][36];
  int ncl = lane & 15, hf = lane >> 4;

  v16bf qa0 = load_a16x32(Q, kDH, qt0, 0, lane);
  v16bf qa1 = load_a16x32(Q, kDH, qt0, 32, lane);

  v8f acc[4];
  float mrow[8], lrow[8];
#pragma unroll
  for (int nt = 0; nt < 4; ++nt)
#pragma unroll
    for (int r = 0; r < 8; ++r) acc[nt][r] = 0.f;
#pragma unroll
  for (int r = 0; r < 8; ++r) { mrow[r] = -1e30f; lrow[r] = 0.f; }

  for (int s0 = 0; s0 <= qt0 + 15; s0 += 32) {
    // S = Q @ K^T : two 16x16 tiles over 32 keys, K-dim = DH = 64
    v8f st0, st1;
#pragma unroll
    for (int r = 0; r < 8; ++r) { st0[r] = 0.f; st1[r] = 0.f; }
    {
      v16bf kb0 = load_b32x16(Kp, kDH, s0,      0,  lane);
      v16bf kb1 = load_b32x16(Kp, kDH, s0 + 16, 0,  lane);
      st0 = wmma_bf16(qa0, kb0, st0);
      st1 = wmma_bf16(qa0, kb1, st1);
      v16bf kb2 = load_b32x16(Kp, kDH, s0,      32, lane);
      v16bf kb3 = load_b32x16(Kp, kDH, s0 + 16, 32, lane);
      st0 = wmma_bf16(qa1, kb2, st0);
      st1 = wmma_bf16(qa1, kb3, st1);
    }
    // online softmax per row (rows split across lane halves per C-layout)
#pragma unroll
    for (int r = 0; r < 8; ++r) {
      int rowt = qt0 + r + hf * 8;
      float c0 = st0[r] * kScale;
      float c1 = st1[r] * kScale;
      if (s0 + ncl      > rowt) c0 = -1e30f;
      if (s0 + 16 + ncl > rowt) c1 = -1e30f;
      float mx = fmaxf(c0, c1);
#pragma unroll
      for (int off = 8; off > 0; off >>= 1) mx = fmaxf(mx, __shfl_xor(mx, off, 32));
      float mnew = fmaxf(mrow[r], mx);
      float sc = __expf(mrow[r] - mnew);
      mrow[r] = mnew;
      float p0 = __expf(c0 - mnew);
      float p1 = __expf(c1 - mnew);
      float sm = p0 + p1;
#pragma unroll
      for (int off = 8; off > 0; off >>= 1) sm += __shfl_xor(sm, off, 32);
      lrow[r] = lrow[r] * sc + sm;
#pragma unroll
      for (int nt = 0; nt < 4; ++nt) acc[nt][r] *= sc;
      st0[r] = p0; st1[r] = p1;
    }
    // stage P (16x32) to this wave's LDS slice; same-wave DS ops are in-order
#pragma unroll
    for (int r = 0; r < 8; ++r) {
      int rr = r + hf * 8;
      Pt[wave][rr][ncl]      = f2bf(st0[r]);
      Pt[wave][rr][16 + ncl] = f2bf(st1[r]);
    }
    asm volatile("" ::: "memory");  // compiler fence: keep store->load order
    v16bf pa = load_a16x32(&Pt[wave][0][0], 36, 0, 0, lane);
    asm volatile("" ::: "memory");
    // acc += P @ V  (K-dim = 32 keys, N = DH in 4 tiles)
#pragma unroll
    for (int nt = 0; nt < 4; ++nt) {
      v16bf vb = load_b32x16(Vt, kT, nt * 16, s0, lane);
      acc[nt] = wmma_bf16(pa, vb, acc[nt]);
    }
  }
  // normalize and write concat-head output [BT, C]
#pragma unroll
  for (int r = 0; r < 8; ++r) {
    float inv = 1.f / lrow[r];
    int t = qt0 + r + hf * 8;
    size_t rowg = (size_t)b * kT + t;
#pragma unroll
    for (int nt = 0; nt < 4; ++nt) {
      int col = h * kDH + nt * 16 + ncl;
      attn_out[rowg * kC + col] = f2bf(acc[nt][r] * inv);
    }
  }
}

// ---------------- generic bf16 GEMM with epilogue ----------------
// out = [relu](A @ Bt^T + bias) [+ res] ; A [M,K] bf16 row-major,
// Bt [N,K] bf16 (pre-transposed). grid=(N/64, M/128), block=128 (4 waves).
// Each wave: 32(M) x 64(N). B tile staged in LDS via async copy (if avail).
template <bool RELU, bool RES, bool OUTBF>
__global__ void k_gemm(const unsigned short* __restrict__ A,
                       const unsigned short* __restrict__ Bt,
                       const float* __restrict__ bias,
                       const float* __restrict__ res,
                       float* __restrict__ outf,
                       unsigned short* __restrict__ outb,
                       int M, int N, int K) {
  int lane = threadIdx.x & 31, wave = threadIdx.x >> 5;
  int row0 = blockIdx.y * 128 + wave * 32;
  int n0b  = blockIdx.x * 64;
  v8f acc[2][4];
#pragma unroll
  for (int mt = 0; mt < 2; ++mt)
#pragma unroll
    for (int nt = 0; nt < 4; ++nt)
#pragma unroll
      for (int r = 0; r < 8; ++r) acc[mt][nt][r] = 0.f;

#if HAS_ASYNC_LDS
  // double-buffered B tile: [2][64 rows][40] (pad: 80B row stride,
  // conflict-free for 16 lanes, 16B aligned)
  __shared__ unsigned short Bl[2][64][40];
  auto issueB = [&](int buf, int k0) {
    // 64x32 bf16 tile = 256 16B-chunks; 128 threads x 2 chunks
    int c0 = wave * 64 + lane;
#pragma unroll
    for (int it = 0; it < 2; ++it) {
      int c = c0 + it * 32;
      int rowb = c >> 2, j = c & 3;
      const unsigned short* src = Bt + (size_t)(n0b + rowb) * K + k0 + j * 8;
      unsigned short* dst = &Bl[buf][rowb][j * 8];
      __builtin_amdgcn_global_load_async_to_lds_b128(
          (as1_v4i)(uintptr_t)src, (as3_v4i)(uint32_t)(uintptr_t)dst, 0, 0);
    }
  };
  int nsteps = K / 32;
  issueB(0, 0);
  for (int i = 0; i < nsteps; ++i) {
    int kc = i * 32;
    if (i + 1 < nsteps) {
      issueB((i + 1) & 1, kc + 32);
      WAIT_ASYNC(2);   // this wave's 2 loads for buffer i complete (in-order)
    } else {
      WAIT_ASYNC(0);
    }
    __syncthreads();
    v16bf a0 = load_a16x32(A, K, row0,      kc, lane);
    v16bf a1 = load_a16x32(A, K, row0 + 16, kc, lane);
    const unsigned short* Bb = &Bl[i & 1][0][0];
#pragma unroll
    for (int nt = 0; nt < 4; ++nt) {
      v16bf b = load_b32x16(Bb, 40, nt * 16, 0, lane);
      acc[0][nt] = wmma_bf16(a0, b, acc[0][nt]);
      acc[1][nt] = wmma_bf16(a1, b, acc[1][nt]);
    }
    __syncthreads();
  }
#else
  for (int k0 = 0; k0 < K; k0 += 32) {
    v16bf a0 = load_a16x32(A, K, row0,      k0, lane);
    v16bf a1 = load_a16x32(A, K, row0 + 16, k0, lane);
#pragma unroll
    for (int nt = 0; nt < 4; ++nt) {
      v16bf b = load_b32x16(Bt, K, n0b + nt * 16, k0, lane);
      acc[0][nt] = wmma_bf16(a0, b, acc[0][nt]);
      acc[1][nt] = wmma_bf16(a1, b, acc[1][nt]);
    }
  }
#endif
  int ncl = lane & 15, hf = lane >> 4;
#pragma unroll
  for (int mt = 0; mt < 2; ++mt)
#pragma unroll
    for (int nt = 0; nt < 4; ++nt) {
      int col = n0b + nt * 16 + ncl;
      float bv = bias[col];
#pragma unroll
      for (int r = 0; r < 8; ++r) {
        int row = row0 + mt * 16 + r + hf * 8;
        float v = acc[mt][nt][r] + bv;
        if (RELU) v = fmaxf(v, 0.f);
        if (RES)  v += res[(size_t)row * N + col];
        if (OUTBF) outb[(size_t)row * N + col] = f2bf(v);
        else       outf[(size_t)row * N + col] = v;
      }
    }
}

// ---------------- host side ----------------
static inline long cdiv(long a, long b) { return (a + b - 1) / b; }

extern "C" void kernel_launch(void* const* d_in, const int* in_sizes, int n_in,
                              void* d_out, int out_size, void* d_ws, size_t ws_size,
                              hipStream_t stream) {
  (void)in_sizes; (void)n_in; (void)out_size; (void)ws_size;
  const float* x   = (const float*)d_in[0];
  const float* Wq  = (const float*)d_in[1];
  const float* Wk  = (const float*)d_in[2];
  const float* Wv  = (const float*)d_in[3];
  const float* Wo  = (const float*)d_in[4];
  const float* bo  = (const float*)d_in[5];
  const float* W1  = (const float*)d_in[6];
  const float* b1  = (const float*)d_in[7];
  const float* W2  = (const float*)d_in[8];
  const float* b2  = (const float*)d_in[9];
  const float* g1  = (const float*)d_in[10];
  const float* be1 = (const float*)d_in[11];
  const float* g2  = (const float*)d_in[12];
  const float* be2 = (const float*)d_in[13];
  float* out = (float*)d_out;

  // carve workspace
  char* w = (char*)d_ws;
  size_t off = 0;
  auto alloc = [&](size_t bytes) {
    void* p = w + off;
    off = (off + bytes + 255) & ~(size_t)255;
    return p;
  };
  unsigned short* hn1   = (unsigned short*)alloc((size_t)kBT * kC * 2);
  unsigned short* hn2   = (unsigned short*)alloc((size_t)kBT * kC * 2);
  unsigned short* attnb = (unsigned short*)alloc((size_t)kBT * kC * 2);
  unsigned short* qb    = (unsigned short*)alloc((size_t)kBT * kC * 2);
  unsigned short* kbuf  = (unsigned short*)alloc((size_t)kBT * kC * 2);
  unsigned short* vtb   = (unsigned short*)alloc((size_t)kBT * kC * 2);
  unsigned short* ff1b  = (unsigned short*)alloc((size_t)kBT * kFF * 2);
  float*          x1    = (float*)         alloc((size_t)kBT * kC * 4);
  unsigned short* Wq_t  = (unsigned short*)alloc((size_t)kH * kC * kDH * 2);
  unsigned short* Wk_t  = (unsigned short*)alloc((size_t)kH * kC * kDH * 2);
  unsigned short* Wv_t  = (unsigned short*)alloc((size_t)kH * kC * kDH * 2);
  unsigned short* Wo_t  = (unsigned short*)alloc((size_t)kC * kC * 2);
  unsigned short* W1_t  = (unsigned short*)alloc((size_t)kC * kFF * 2);
  unsigned short* W2_t  = (unsigned short*)alloc((size_t)kFF * kC * 2);

  // 1) weight transpose + bf16 convert
  {
    long t1 = (long)kH * kC * kDH;
    k_transpose_bf16<<<cdiv(t1, 256), 256, 0, stream>>>(Wq, Wq_t, kC, kDH, t1);
    k_transpose_bf16<<<cdiv(t1, 256), 256, 0, stream>>>(Wk, Wk_t, kC, kDH, t1);
    k_transpose_bf16<<<cdiv(t1, 256), 256, 0, stream>>>(Wv, Wv_t, kC, kDH, t1);
    long t2 = (long)kC * kC;
    k_transpose_bf16<<<cdiv(t2, 256), 256, 0, stream>>>(Wo, Wo_t, kC, kC, t2);
    long t3 = (long)kC * kFF;
    k_transpose_bf16<<<cdiv(t3, 256), 256, 0, stream>>>(W1, W1_t, kC, kFF, t3);
    k_transpose_bf16<<<cdiv(t3, 256), 256, 0, stream>>>(W2, W2_t, kFF, kC, t3);
  }
  // 2) LN1
  k_layernorm<<<kBT, 256, 0, stream>>>(x, g1, be1, hn1);
  // 3) QKV projections
  k_qkv<<<dim3(kBT / 128, kH, 3), 128, 0, stream>>>(hn1, Wq_t, Wk_t, Wv_t,
                                                    qb, kbuf, vtb);
  // 4) causal flash attention
  k_attn<<<dim3(kT / 64, kB * kH), 128, 0, stream>>>(qb, kbuf, vtb, attnb);
  // 5) output projection + bias + residual -> x1 (f32)
  k_gemm<false, true, false><<<dim3(kC / 64, kBT / 128), 128, 0, stream>>>(
      attnb, Wo_t, bo, x, x1, nullptr, kBT, kC, kC);
  // 6) LN2
  k_layernorm<<<kBT, 256, 0, stream>>>(x1, g2, be2, hn2);
  // 7) FF1 + bias + relu -> bf16
  k_gemm<true, false, true><<<dim3(kFF / 64, kBT / 128), 128, 0, stream>>>(
      hn2, W1_t, b1, nullptr, nullptr, ff1b, kBT, kFF, kC);
  // 8) FF2 + bias + residual -> out (f32)
  k_gemm<false, true, false><<<dim3(kC / 64, kBT / 128), 128, 0, stream>>>(
      ff1b, W2_t, b2, x1, out, nullptr, kBT, kC, kFF);
}
